// FrequencyAttention_42056319762483
// MI455X (gfx1250) — matrix operand order repaired and checked by reading, not compiled, not run on hardware
//
#include <hip/hip_runtime.h>
#include <hip/hip_bf16.h>

// ---------------------------------------------------------------------------
// FrequencyAttention on MI455X (gfx1250, wave32, WMMA).
//   K0: weights fp32 -> f16 in ws, zero BN accumulators
//   K1: fused qkv GEMM + attention + out-proj + residual (v_wmma_f32_16x16x32_f16)
//       writes pre-BN y to d_out, accumulates per-channel sum/sumsq atomics.
//       Loop unrolling capped so the wave stays within 256 VGPRs (no
//       s_set_vgpr_msb churn, better occupancy).
//   K2: BN normalize in place (float4, memory bound)
// ---------------------------------------------------------------------------

typedef __attribute__((ext_vector_type(16))) _Float16 v16h;
typedef __attribute__((ext_vector_type(8)))  _Float16 v8h;
typedef __attribute__((ext_vector_type(8)))  float    v8f;

#define NUM_BINS  32
#define CHANNELS  128
#define NUM_HEADS 4
#define HEAD_DIM  32
#define BATCH     8192
#define ATT_SCALE 0.17677669529663687f   // 1/sqrt(32)
#define BN_EPS    1e-5f
#define BPB       4                      // batches per (single-wave) block

#define LPQ 40    // LDS pitch (halfs) for qT/kT/vT/pT : 80B, 16B aligned, bank-staggered
#define LPS 36    // LDS pitch (floats) for S
#define LPA 136   // LDS pitch (halfs) for attention-out (272B, 16B aligned)

__device__ __forceinline__ v8f wmma16(v16h a, v16h b, v8f c) {
  return __builtin_amdgcn_wmma_f32_16x16x32_f16(false, a, false, b, (short)0, c,
                                                false, false);
}

// A-fragment (16x32 f16, row-major, pitch 128): lane = row lo, elements
// K = kb + hf*16 + e (contiguous, matches B fragments)
__device__ __forceinline__ v16h load_a_f16(const _Float16* row, int kb, int hf) {
  v8h c0 = *(const v8h*)(row + kb + hf * 16);
  v8h c1 = *(const v8h*)(row + kb + hf * 16 + 8);
  v16h a;
#pragma unroll
  for (int i = 0; i < 8; ++i) { a[i] = c0[i]; a[i + 8] = c1[i]; }
  return a;
}

// B-fragment (32x16 f16): lane = column (lo), elements K = hf*16 + e, i.e.
// 16 contiguous halfs starting at p (caller adds col*pitch + hf*16).
__device__ __forceinline__ v16h load_b_f16(const _Float16* p) {
  v8h c0 = *(const v8h*)(p);
  v8h c1 = *(const v8h*)(p + 8);
  v16h b;
#pragma unroll
  for (int i = 0; i < 8; ++i) { b[i] = c0[i]; b[i + 8] = c1[i]; }
  return b;
}

// --------------------------------------------------------------------------
// K0: weight conversion + accumulator zeroing
// --------------------------------------------------------------------------
__global__ void fa_prep(const float* __restrict__ qkvw, const float* __restrict__ outw,
                        _Float16* __restrict__ w16, _Float16* __restrict__ ow16,
                        float* __restrict__ accum) {
  const int idx = blockIdx.x * blockDim.x + threadIdx.x;
  const int stride = gridDim.x * blockDim.x;
  if (blockIdx.x == 0 && threadIdx.x < 256) accum[threadIdx.x] = 0.0f;
  for (int i = idx; i < 3 * CHANNELS * CHANNELS; i += stride)
    w16[i] = (_Float16)qkvw[i];
  for (int i = idx; i < CHANNELS * CHANNELS; i += stride)
    ow16[i] = (_Float16)outw[i];
}

// --------------------------------------------------------------------------
// K1: fused attention block, one wave per workgroup, BPB batches per block
// --------------------------------------------------------------------------
__global__ __launch_bounds__(32) void fa_main(
    const float* __restrict__ x, const float* __restrict__ qkvb,
    const float* __restrict__ outb, const float* __restrict__ emb,
    const _Float16* __restrict__ w16, const _Float16* __restrict__ ow16,
    float* __restrict__ gsum, float* __restrict__ gsq,
    float* __restrict__ out) {
  __shared__ _Float16 qT[32 * LPQ];   // q[n][d]
  __shared__ _Float16 kT[32 * LPQ];   // k[m][d]
  __shared__ _Float16 vT[32 * LPQ];   // v^T[d][m]
  __shared__ _Float16 pT[32 * LPQ];   // softmax P[n][m]
  __shared__ float    sT[32 * LPS];   // scores S[n][m]
  __shared__ _Float16 aoT[32 * LPA];  // attn out [n][c=128]
  __shared__ float lsum[CHANNELS], lsq[CHANNELS];
  __shared__ float sQB[3 * CHANNELS], sOB[CHANNELS], sEmb[252];

  const int lane = threadIdx.x;
  const int hf = lane >> 4;        // lane-half
  const int lo = lane & 15;

  for (int i = lane; i < CHANNELS; i += 32) { lsum[i] = 0.f; lsq[i] = 0.f; sOB[i] = outb[i]; }
  for (int i = lane; i < 3 * CHANNELS; i += 32) sQB[i] = qkvb[i];
  for (int i = lane; i < 252; i += 32) sEmb[i] = emb[i];

  for (int bi = 0; bi < BPB; ++bi) {
    const int b = blockIdx.x * BPB + bi;
    const float* xb = x + (size_t)b * (CHANNELS * NUM_BINS);
    float* yb = out + (size_t)b * (CHANNELS * NUM_BINS);

    // x as B-fragments for qkv GEMM: B[c][n], lane col n, elements over c.
    v16h bx[4][2];
#pragma unroll
    for (int kc = 0; kc < 4; ++kc)
#pragma unroll
      for (int nc = 0; nc < 2; ++nc) {
        const int n = nc * 16 + lo;
#pragma unroll
        for (int e = 0; e < 16; ++e) {
          const int c = kc * 32 + hf * 16 + e;
          bx[kc][nc][e] = (_Float16)xb[c * NUM_BINS + n];
        }
      }

    for (int h = 0; h < NUM_HEADS; ++h) {
      // ---- q/k/v projection for this head: rows R..R+31 of qkv_w ----
      // t-loop deliberately NOT unrolled: caps live accumulators at 4 tiles
      // (32 VGPRs) + transient A-fragments, keeping the wave under 256 VGPRs.
#pragma unroll 1
      for (int t = 0; t < 3; ++t) {
        const int R = t * CHANNELS + h * HEAD_DIM;
        // prefetch next weight panel while this one computes
        __builtin_prefetch(w16 + (size_t)(((t + 1) % 3) * CHANNELS + h * HEAD_DIM + lo) * CHANNELS, 0, 3);
#pragma unroll
        for (int mt = 0; mt < 2; ++mt) {
          const _Float16* wrow = w16 + (size_t)(R + mt * 16 + lo) * CHANNELS;
#pragma unroll
          for (int nt = 0; nt < 2; ++nt) {
            v8f acc = {0.f, 0.f, 0.f, 0.f, 0.f, 0.f, 0.f, 0.f};
#pragma unroll
            for (int kc = 0; kc < 4; ++kc)
              acc = wmma16(load_a_f16(wrow, kc * 32, hf), bx[kc][nt], acc);
            if (t < 2) {                    // q,k: packed b128 store, [pos][d]
              _Float16* dst = t ? kT : qT;
              v8h pk;
#pragma unroll
              for (int r = 0; r < 8; ++r)
                pk[r] = (_Float16)(acc[r] + sQB[R + mt * 16 + r + 8 * hf]);
              *(v8h*)(dst + (nt * 16 + lo) * LPQ + mt * 16 + hf * 8) = pk;
            } else {                        // v: scatter-transpose -> vT[d][m]
#pragma unroll
              for (int r = 0; r < 8; ++r) {
                const int d = mt * 16 + r + 8 * hf;
                vT[d * LPQ + nt * 16 + lo] = (_Float16)(acc[r] + sQB[R + d]);
              }
            }
          }
        }
      }
      // ---- S = q k^T (M=n, N=m, K=d=32 -> one WMMA per 16x16 tile) ----
#pragma unroll
      for (int mt = 0; mt < 2; ++mt)
#pragma unroll
        for (int nt = 0; nt < 2; ++nt) {
          v16h a = load_a_f16(qT + (mt * 16 + lo) * LPQ, 0, hf);
          v16h bf = load_b_f16(kT + (nt * 16 + lo) * LPQ + hf * 16);
          v8f acc = {0.f, 0.f, 0.f, 0.f, 0.f, 0.f, 0.f, 0.f};
          acc = wmma16(a, bf, acc);
#pragma unroll
          for (int r = 0; r < 8; ++r)
            sT[(mt * 16 + r + 8 * hf) * LPS + nt * 16 + lo] = acc[r];
        }
      // ---- softmax: lane = row n, rel-pos bias + scale fused ----
      {
        const int n = lane;
        float vals[32];
        float mx = -3.4e38f;
#pragma unroll
        for (int j = 0; j < 32; ++j) {
          int idx = j - n + NUM_BINS - 1;
          idx = idx < 0 ? 0 : (idx > 62 ? 62 : idx);
          float s = sT[n * LPS + j] * ATT_SCALE + sEmb[idx * NUM_HEADS + h];
          vals[j] = s;
          mx = fmaxf(mx, s);
        }
        float sum = 0.f;
#pragma unroll
        for (int j = 0; j < 32; ++j) { float e = __expf(vals[j] - mx); vals[j] = e; sum += e; }
        const float inv = 1.f / sum;
#pragma unroll
        for (int j = 0; j < 32; ++j) pT[n * LPQ + j] = (_Float16)(vals[j] * inv);
      }
      // ---- O = P V (M=n, N=d, K=m=32) -> aoT[n][h*32+d] ----
#pragma unroll
      for (int mt = 0; mt < 2; ++mt)
#pragma unroll
        for (int nt = 0; nt < 2; ++nt) {
          v16h a = load_a_f16(pT + (mt * 16 + lo) * LPQ, 0, hf);
          v16h bf = load_b_f16(vT + (nt * 16 + lo) * LPQ + hf * 16);
          v8f acc = {0.f, 0.f, 0.f, 0.f, 0.f, 0.f, 0.f, 0.f};
          acc = wmma16(a, bf, acc);
#pragma unroll
          for (int r = 0; r < 8; ++r) {
            const int n2 = mt * 16 + r + 8 * hf;
            aoT[n2 * LPA + h * HEAD_DIM + nt * 16 + lo] = (_Float16)acc[r];
          }
        }
    }  // heads

    // ---- output projection: y = out_w @ ao + out_b + x ----
    v16h ba[4][2];
#pragma unroll
    for (int kc = 0; kc < 4; ++kc)
#pragma unroll
      for (int nc = 0; nc < 2; ++nc)
        ba[kc][nc] = load_b_f16(aoT + (nc * 16 + lo) * LPA + kc * 32 + hf * 16);

    // mt-loop NOT unrolled: 2 live accumulator tiles + 4 A-fragments max.
#pragma unroll 1
    for (int mt = 0; mt < 8; ++mt) {
      const _Float16* wrow = ow16 + (size_t)(mt * 16 + lo) * CHANNELS;
      __builtin_prefetch(ow16 + (size_t)(((mt + 1) & 7) * 16 + lo) * CHANNELS, 0, 3);
#pragma unroll
      for (int nc = 0; nc < 2; ++nc) {
        v8f acc = {0.f, 0.f, 0.f, 0.f, 0.f, 0.f, 0.f, 0.f};
#pragma unroll
        for (int kc = 0; kc < 4; ++kc)
          acc = wmma16(load_a_f16(wrow, kc * 32, hf), ba[kc][nc], acc);
#pragma unroll
        for (int r = 0; r < 8; ++r) {
          const int c = mt * 16 + r + 8 * hf;
          const int n = nc * 16 + lo;
          const float y = acc[r] + sOB[c] + xb[c * NUM_BINS + n];
          yb[c * NUM_BINS + n] = y;
          float s1 = y, s2 = y * y;
#pragma unroll
          for (int off = 1; off < 16; off <<= 1) {
            s1 += __shfl_xor(s1, off, 32);
            s2 += __shfl_xor(s2, off, 32);
          }
          if (lo == 0) { lsum[c] += s1; lsq[c] += s2; }  // lanes 0,16 -> distinct c
        }
      }
    }
  }  // batch loop

  for (int i = lane; i < CHANNELS; i += 32) {
    atomicAdd(&gsum[i], lsum[i]);
    atomicAdd(&gsq[i], lsq[i]);
  }
}

// --------------------------------------------------------------------------
// K2: BatchNorm apply (in-place on d_out), float4 per thread
// --------------------------------------------------------------------------
__global__ void fa_bn_apply(float* __restrict__ out, const float* __restrict__ gsum,
                            const float* __restrict__ gsq,
                            const float* __restrict__ gamma,
                            const float* __restrict__ beta) {
  const size_t i4 = ((size_t)blockIdx.x * blockDim.x + threadIdx.x) * 4;
  const size_t total = (size_t)BATCH * CHANNELS * NUM_BINS;
  if (i4 >= total) return;
  const int c = (int)((i4 >> 5) & (CHANNELS - 1));
  const float cnt = (float)BATCH * (float)NUM_BINS;
  const float mean = gsum[c] / cnt;
  const float var = gsq[c] / cnt - mean * mean;
  const float inv = rsqrtf(var + BN_EPS);
  const float sc = gamma[c] * inv;
  const float sh = beta[c] - mean * sc;
  float4 y = *(float4*)(out + i4);
  y.x = y.x * sc + sh;
  y.y = y.y * sc + sh;
  y.z = y.z * sc + sh;
  y.w = y.w * sc + sh;
  *(float4*)(out + i4) = y;
}

// --------------------------------------------------------------------------
extern "C" void kernel_launch(void* const* d_in, const int* in_sizes, int n_in,
                              void* d_out, int out_size, void* d_ws, size_t ws_size,
                              hipStream_t stream) {
  (void)in_sizes; (void)n_in; (void)out_size; (void)ws_size;
  const float* x    = (const float*)d_in[0];
  const float* qkvw = (const float*)d_in[1];
  const float* qkvb = (const float*)d_in[2];
  const float* outw = (const float*)d_in[3];
  const float* outb = (const float*)d_in[4];
  const float* emb  = (const float*)d_in[5];
  const float* gam  = (const float*)d_in[6];
  const float* bet  = (const float*)d_in[7];
  float* out = (float*)d_out;

  char* ws = (char*)d_ws;
  float*    gsum = (float*)ws;                                   // 128 f32
  float*    gsq  = gsum + CHANNELS;                              // 128 f32
  _Float16* w16  = (_Float16*)(ws + 1024);                       // 384x128 f16
  _Float16* ow16 = (_Float16*)(ws + 1024 + 3 * CHANNELS * CHANNELS * 2);  // 128x128 f16

  fa_prep<<<64, 256, 0, stream>>>(qkvw, outw, w16, ow16, gsum);
  fa_main<<<BATCH / BPB, 32, 0, stream>>>(x, qkvb, outb, emb, w16, ow16, gsum, gsq, out);
  const size_t total = (size_t)BATCH * CHANNELS * NUM_BINS;
  fa_bn_apply<<<(unsigned)((total / 4 + 255) / 256), 256, 0, stream>>>(out, gsum, gsq, gam, bet);
}
